// WordLevelAlignmentModule_40870908789485
// MI455X (gfx1250) — matrix operand order repaired and checked by reading, not compiled, not run on hardware
//
#include <hip/hip_runtime.h>
#include <hip/hip_bf16.h>
#include <math.h>

typedef __attribute__((ext_vector_type(16))) _Float16 v16h;
typedef __attribute__((ext_vector_type(8)))  float    v8f;
typedef __attribute__((ext_vector_type(4)))  float    f4;   // builtin vec, AS-safe

// global (address space 1) pointer types: force global_load/global_store
// instead of flat_* so VMEM traffic only touches LOADcnt/STOREcnt (not DScnt)
typedef const __attribute__((address_space(1))) f4*    gf4cp;
typedef const __attribute__((address_space(1))) float* gfcp;
typedef       __attribute__((address_space(1))) float* gfp;

#define Bk 16
#define Tk 256
#define Ak 3000
#define APk 3008          // A padded to multiple of 32 for guard-free WMMA GEMMs
#define Hk 768
#define NHk 4
#define HDk 192

__device__ __forceinline__ v8f zero8() {
    v8f z = {0.f,0.f,0.f,0.f,0.f,0.f,0.f,0.f};
    return z;
}

__device__ __forceinline__ float wave_red_sum(float v) {
    #pragma unroll
    for (int o = 16; o > 0; o >>= 1) v += __shfl_down(v, o, 32);
    return v;
}
__device__ __forceinline__ float wave_red_max(float v) {
    #pragma unroll
    for (int o = 16; o > 0; o >>= 1) v = fmaxf(v, __shfl_down(v, o, 32));
    return v;
}

// block reduce across 8 waves (blockDim.x == 256)
__device__ __forceinline__ float block_red(float v, float* sm, bool domax) {
    const int lane = threadIdx.x & 31;
    const int w    = threadIdx.x >> 5;
    const int nw   = blockDim.x >> 5;
    v = domax ? wave_red_max(v) : wave_red_sum(v);
    __syncthreads();
    if (lane == 0) sm[w] = v;
    __syncthreads();
    float r = domax ? -__builtin_inff() : 0.f;
    for (int i = 0; i < nw; ++i) r = domax ? fmaxf(r, sm[i]) : (r + sm[i]);
    return r;
}

__device__ __forceinline__ void pack_a(v16h& a, const float* p) {
    const f4 p0 = ((gf4cp)(p + 0))[0];
    const f4 p1 = ((gf4cp)(p + 4))[0];
    const f4 p2 = ((gf4cp)(p + 8))[0];
    const f4 p3 = ((gf4cp)(p + 12))[0];
    a[0]=(_Float16)p0.x;  a[1]=(_Float16)p0.y;  a[2]=(_Float16)p0.z;  a[3]=(_Float16)p0.w;
    a[4]=(_Float16)p1.x;  a[5]=(_Float16)p1.y;  a[6]=(_Float16)p1.z;  a[7]=(_Float16)p1.w;
    a[8]=(_Float16)p2.x;  a[9]=(_Float16)p2.y;  a[10]=(_Float16)p2.z; a[11]=(_Float16)p2.w;
    a[12]=(_Float16)p3.x; a[13]=(_Float16)p3.y; a[14]=(_Float16)p3.z; a[15]=(_Float16)p3.w;
}

__device__ __forceinline__ void pack_b_vec(v16h& b, const float* p) {
    const f4 q0 = ((gf4cp)(p + 0))[0];
    const f4 q1 = ((gf4cp)(p + 4))[0];
    const f4 q2 = ((gf4cp)(p + 8))[0];
    const f4 q3 = ((gf4cp)(p + 12))[0];
    b[0]=(_Float16)q0.x;  b[1]=(_Float16)q0.y;  b[2]=(_Float16)q0.z;  b[3]=(_Float16)q0.w;
    b[4]=(_Float16)q1.x;  b[5]=(_Float16)q1.y;  b[6]=(_Float16)q1.z;  b[7]=(_Float16)q1.w;
    b[8]=(_Float16)q2.x;  b[9]=(_Float16)q2.y;  b[10]=(_Float16)q2.z; b[11]=(_Float16)q2.w;
    b[12]=(_Float16)q3.x; b[13]=(_Float16)q3.y; b[14]=(_Float16)q3.z; b[15]=(_Float16)q3.w;
}

// -----------------------------------------------------------------------------
// Branch-free batched WMMA GEMM:  C = alpha * X @ W^T (+bias) (+resid) (relu)
// Requirements (guaranteed by host): M % 32 == 0, N % 64 == 0, K % 32 == 0.
//   X element (m,k) at Xbase + m*ldx + k
//   W element (n,k) at Wbase + n*wns + k*wks   (BVEC => wks==1, vector loads)
//   C element (m,n) at Cbase + m*ldc + n
//   batch e: outer = e/inner_n, inner = e%inner_n; base = outer*s?_o + inner*s?_i
// Wave tile: 32x64 -> 2 A-frags x 4 B-frags, 8 v_wmma_f32_16x16x32_f16 / chunk.
// flags: bit0 = add resid (indexed like C), bit1 = relu
// -----------------------------------------------------------------------------
template<bool BVEC>
__global__ void __launch_bounds__(256)
gemm_wmma_kernel(const float* __restrict__ X,
                 const float* __restrict__ W,
                 const float* __restrict__ bias,
                 const float* __restrict__ resid,
                 float* __restrict__ C,
                 int M, int N, int K,
                 int ldx, long long wns, long long wks, int ldc,
                 long long sx_o, long long sx_i,
                 long long sw_o, long long sw_i,
                 long long sc_o, long long sc_i,
                 int nbatch, int inner_n,
                 float alpha, int flags)
{
    const int lane = threadIdx.x & 31;
    const int half = lane >> 4;
    const int l16  = lane & 15;

    const int tilesM = M >> 5;
    const int tilesN = N >> 6;
    const long long tmn   = (long long)tilesM * tilesN;
    const long long total = tmn * nbatch;

    const int wavesPerBlock = blockDim.x >> 5;
    long long wid = (long long)blockIdx.x * wavesPerBlock + (threadIdx.x >> 5);
    const long long wstride = (long long)gridDim.x * wavesPerBlock;

    for (; wid < total; wid += wstride) {
        const int e     = (int)(wid / tmn);
        const int tile  = (int)(wid % tmn);
        const int tm    = tile / tilesN;
        const int tn    = tile % tilesN;
        const int outer = e / inner_n;
        const int inner = e % inner_n;

        const float* Xb = X + (long long)outer * sx_o + (long long)inner * sx_i;
        const float* Wb = W + (long long)outer * sw_o + (long long)inner * sw_i;
        float*       Cb = C + (long long)outer * sc_o + (long long)inner * sc_i;
        const float* Rb = resid ? (resid + (long long)outer * sc_o + (long long)inner * sc_i)
                                : nullptr;

        const int m0 = tm << 5;
        const int n0 = tn << 6;

        v8f acc[2][4];
        #pragma unroll
        for (int i = 0; i < 2; ++i)
            #pragma unroll
            for (int j = 0; j < 4; ++j) acc[i][j] = zero8();

        // per-lane streaming pointers (hoisted; advance by pointer adds)
        const float* aptr[2];
        aptr[0] = Xb + (long long)(m0 + l16)      * ldx + half * 16;
        aptr[1] = Xb + (long long)(m0 + 16 + l16) * ldx + half * 16;
        const float* bptr[4];
        #pragma unroll
        for (int j = 0; j < 4; ++j)
            bptr[j] = Wb + (long long)(n0 + j * 16 + l16) * wns
                         + (long long)(half * 16) * wks;

        const int nchunks = K >> 5;
        for (int kc = 0; kc < nchunks; ++kc) {
            v16h af[2], bf[4];
            #pragma unroll
            for (int i = 0; i < 2; ++i) {
                pack_a(af[i], aptr[i]);
                __builtin_prefetch((const void*)(gfcp)(aptr[i] + 32), 0, 1);
                aptr[i] += 32;
            }
            if (BVEC) {
                #pragma unroll
                for (int j = 0; j < 4; ++j) {
                    pack_b_vec(bf[j], bptr[j]);
                    __builtin_prefetch((const void*)(gfcp)(bptr[j] + 32), 0, 1);
                    bptr[j] += 32;
                }
            } else {
                #pragma unroll
                for (int j = 0; j < 4; ++j) {
                    #pragma unroll
                    for (int u = 0; u < 16; ++u)
                        bf[j][u] = (_Float16)((gfcp)bptr[j])[(long long)u * wks];
                    bptr[j] += 32 * wks;
                }
            }
            #pragma unroll
            for (int i = 0; i < 2; ++i)
                #pragma unroll
                for (int j = 0; j < 4; ++j)
                    acc[i][j] = __builtin_amdgcn_wmma_f32_16x16x32_f16(
                        false, af[i], false, bf[j], (short)0, acc[i][j], false, false);
        }

        // epilogue: C layout lane = N column, M = half*8 + r
        #pragma unroll
        for (int i = 0; i < 2; ++i) {
            const int mbase = m0 + i * 16 + half * 8;
            #pragma unroll
            for (int j = 0; j < 4; ++j) {
                const int nn = n0 + j * 16 + l16;
                const float bv = bias ? ((gfcp)bias)[nn] : 0.f;
                #pragma unroll
                for (int r = 0; r < 8; ++r) {
                    const int mm = mbase + r;
                    float v = alpha * acc[i][j][r] + bv;
                    const long long idx = (long long)mm * ldc + nn;
                    if (flags & 1) v += ((gfcp)Rb)[idx];
                    if (flags & 2) v = fmaxf(v, 0.f);
                    ((gfp)Cb)[idx] = v;
                }
            }
        }
    }
}

// -----------------------------------------------------------------------------
// Masked softmax over A per (b,h,t) row (ld = APk), in place, zeroing the pad
// columns [Ak, APk), plus head-mean -> align_out (ld = Ak).
// One block (256 threads) per (b,t) row; loops over the 4 heads.
// -----------------------------------------------------------------------------
__global__ void __launch_bounds__(256)
softmax_mean_kernel(float* __restrict__ attn,            // [B,NH,T,APk]
                    const float* __restrict__ amask,     // [B,A]
                    float* __restrict__ align_out)       // [B,T,A]
{
    __shared__ float sm[8];
    const int bt  = blockIdx.x;
    const int b   = bt / Tk;
    const int t   = bt % Tk;
    const int tid = threadIdx.x;

    float macc[12];
    #pragma unroll
    for (int i = 0; i < 12; ++i) macc[i] = 0.f;

    for (int h = 0; h < NHk; ++h) {
        float* row = attn + (((long long)(b * NHk + h)) * Tk + t) * APk;

        float mx = -__builtin_inff();
        #pragma unroll
        for (int i = 0; i < 12; ++i) {
            const int s = tid + i * 256;
            if (s < Ak) {
                const float m = ((gfcp)amask)[b * Ak + s];
                if (m > 0.f) mx = fmaxf(mx, ((gfcp)row)[s]);
            }
        }
        mx = block_red(mx, sm, true);

        float pv[12];
        float sum = 0.f;
        #pragma unroll
        for (int i = 0; i < 12; ++i) {
            const int s = tid + i * 256;
            pv[i] = 0.f;
            if (s < Ak) {
                const float m = ((gfcp)amask)[b * Ak + s];
                if (m > 0.f) { pv[i] = expf(((gfcp)row)[s] - mx); sum += pv[i]; }
            }
        }
        sum = block_red(sum, sm, false);
        const float inv = 1.f / sum;

        #pragma unroll
        for (int i = 0; i < 12; ++i) {
            const int s = tid + i * 256;
            if (s < Ak) {
                const float p = pv[i] * inv;
                ((gfp)row)[s] = p;
                macc[i] += p * (1.f / NHk);
            }
        }
        // zero the pad columns so the ctx GEMM (K = APk) accumulates nothing
        if (tid < APk - Ak) ((gfp)row)[Ak + tid] = 0.f;
    }

    float* arow = align_out + (long long)bt * Ak;
    #pragma unroll
    for (int i = 0; i < 12; ++i) {
        const int s = tid + i * 256;
        if (s < Ak) ((gfp)arow)[s] = macc[i];
    }
}

// -----------------------------------------------------------------------------
// LayerNorm over H=768 per row; 256 threads -> 3 elements each
// -----------------------------------------------------------------------------
__global__ void __launch_bounds__(256)
layernorm_kernel(const float* __restrict__ ypre, const float* __restrict__ g,
                 const float* __restrict__ beta, float* __restrict__ yout)
{
    __shared__ float sm[8];
    const int row = blockIdx.x;
    const int tid = threadIdx.x;
    const float* x = ypre + (long long)row * Hk;

    const float v0 = ((gfcp)x)[tid], v1 = ((gfcp)x)[tid + 256], v2 = ((gfcp)x)[tid + 512];
    float s = block_red(v0 + v1 + v2, sm, false);
    const float mu = s * (1.f / Hk);
    const float d0 = v0 - mu, d1 = v1 - mu, d2 = v2 - mu;
    float sq = block_red(d0 * d0 + d1 * d1 + d2 * d2, sm, false);
    const float rstd = rsqrtf(sq * (1.f / Hk) + 1e-5f);

    float* o = yout + (long long)row * Hk;
    ((gfp)o)[tid]       = d0 * rstd * ((gfcp)g)[tid]       + ((gfcp)beta)[tid];
    ((gfp)o)[tid + 256] = d1 * rstd * ((gfcp)g)[tid + 256] + ((gfcp)beta)[tid + 256];
    ((gfp)o)[tid + 512] = d2 * rstd * ((gfcp)g)[tid + 512] + ((gfcp)beta)[tid + 512];
}

// -----------------------------------------------------------------------------
// Confidence head final dot: out[row] = (h[row,:] . Wc2 + bc2) * tmask[row]
// One wave per row; 8 rows per block.
// -----------------------------------------------------------------------------
__global__ void __launch_bounds__(256)
conf_dot_kernel(const float* __restrict__ hbuf, const float* __restrict__ Wc2,
                const float* __restrict__ bc2, const float* __restrict__ tmask,
                float* __restrict__ out)
{
    const int w    = threadIdx.x >> 5;
    const int lane = threadIdx.x & 31;
    const int row  = blockIdx.x * 8 + w;
    if (row >= Bk * Tk) return;
    const float* hr = hbuf + (long long)row * (Hk / 2);
    float s = 0.f;
    #pragma unroll
    for (int i = 0; i < 12; ++i)
        s += ((gfcp)hr)[lane + 32 * i] * ((gfcp)Wc2)[lane + 32 * i];
    s = wave_red_sum(s);
    if (lane == 0) ((gfp)out)[row] = (s + ((gfcp)bc2)[0]) * ((gfcp)tmask)[row];
}

// -----------------------------------------------------------------------------
// Host side
// -----------------------------------------------------------------------------
static void run_gemm(hipStream_t st, bool bvec, const float* X, const float* W,
                     const float* bias, const float* resid, float* C,
                     int M, int N, int K,
                     int ldx, long long wns, long long wks, int ldc,
                     long long sxo, long long sxi, long long swo, long long swi,
                     long long sco, long long sci, int nb, int innern,
                     float alpha, int flags)
{
    const long long tiles = (long long)(M / 32) * (N / 64) * nb;
    const int wpb = 8;
    long long blocks = (tiles + wpb - 1) / wpb;
    if (blocks > 32768) blocks = 32768;
    if (bvec)
        hipLaunchKernelGGL((gemm_wmma_kernel<true>), dim3((unsigned)blocks), dim3(256), 0, st,
                           X, W, bias, resid, C, M, N, K,
                           ldx, wns, wks, ldc, sxo, sxi, swo, swi, sco, sci,
                           nb, innern, alpha, flags);
    else
        hipLaunchKernelGGL((gemm_wmma_kernel<false>), dim3((unsigned)blocks), dim3(256), 0, st,
                           X, W, bias, resid, C, M, N, K,
                           ldx, wns, wks, ldc, sxo, sxi, swo, swi, sco, sci,
                           nb, innern, alpha, flags);
}

extern "C" void kernel_launch(void* const* d_in, const int* in_sizes, int n_in,
                              void* d_out, int out_size, void* d_ws, size_t ws_size,
                              hipStream_t stream)
{
    (void)in_sizes; (void)n_in; (void)out_size; (void)ws_size;
    const float* text  = (const float*)d_in[0];
    const float* audio = (const float*)d_in[1];
    const float* tmask = (const float*)d_in[2];
    const float* amask = (const float*)d_in[3];
    const float* Wt    = (const float*)d_in[4];
    const float* bt    = (const float*)d_in[5];
    const float* Wa    = (const float*)d_in[6];
    const float* ba    = (const float*)d_in[7];
    const float* ipw   = (const float*)d_in[8];
    const float* ipb   = (const float*)d_in[9];
    const float* outw  = (const float*)d_in[10];
    const float* outb  = (const float*)d_in[11];
    const float* Wo    = (const float*)d_in[12];
    const float* bo    = (const float*)d_in[13];
    const float* lng   = (const float*)d_in[14];
    const float* lnb   = (const float*)d_in[15];
    const float* Wc1   = (const float*)d_in[16];
    const float* bc1   = (const float*)d_in[17];
    const float* Wc2   = (const float*)d_in[18];
    const float* bc2   = (const float*)d_in[19];

    float* ws = (float*)d_ws;

    const size_t BT    = (size_t)Bk * Tk;            // 4096
    const size_t BA    = (size_t)Bk * Ak;            // 48000
    const size_t SLACK = (size_t)(APk - Ak) * Hk;    // 8 rows: pad-column reads
    const size_t TP  = 0;                            // tp  [4096,768]
    const size_t AP  = TP + BT * Hk;                 // ap  [48000,768]
    const size_t QO  = AP + BA * Hk;                 // q   [4096,768]
    const size_t KO  = QO + BT * Hk;                 // k   [48000,768] + slack
    const size_t VO  = KO + BA * Hk + SLACK;         // v   [48000,768] + slack
    const size_t AT  = VO + BA * Hk + SLACK;         // attn [B,NH,T,APk]
    const size_t CTX = TP;                           // reuse tp (dead after q)
    const size_t AO  = AP;                           // reuse ap (dead after k,v)
    const size_t YP  = AP + BT * Hk;                 // pre-LN y
    const size_t HB  = AP + 2 * BT * Hk;             // h buffer [4096,384]

    float* yout   = (float*)d_out;                   // [4096,768]
    float* ascore = yout + BT * Hk;                  // [4096]
    float* amat   = ascore + BT;                     // [B,T,A]

    // 1) tp = text @ Wt^T + bt
    run_gemm(stream, true, text, Wt, bt, nullptr, ws + TP, (int)BT, Hk, Hk,
             Hk, Hk, 1, Hk, 0,0,0,0,0,0, 1,1, 1.f, 0);
    // 2) ap = audio @ Wa^T + ba
    run_gemm(stream, true, audio, Wa, ba, nullptr, ws + AP, (int)BA, Hk, Hk,
             Hk, Hk, 1, Hk, 0,0,0,0,0,0, 1,1, 1.f, 0);
    // 3) q / k / v
    run_gemm(stream, true, ws + TP, ipw,               ipb,          nullptr, ws + QO,
             (int)BT, Hk, Hk, Hk, Hk, 1, Hk, 0,0,0,0,0,0, 1,1, 1.f, 0);
    run_gemm(stream, true, ws + AP, ipw + Hk * Hk,     ipb + Hk,     nullptr, ws + KO,
             (int)BA, Hk, Hk, Hk, Hk, 1, Hk, 0,0,0,0,0,0, 1,1, 1.f, 0);
    run_gemm(stream, true, ws + AP, ipw + 2 * Hk * Hk, ipb + 2 * Hk, nullptr, ws + VO,
             (int)BA, Hk, Hk, Hk, Hk, 1, Hk, 0,0,0,0,0,0, 1,1, 1.f, 0);
    // 4) scores = q @ k^T / sqrt(HD); 64 batches (b,h), inner = NH; N padded
    run_gemm(stream, true, ws + QO, ws + KO, nullptr, nullptr, ws + AT,
             Tk, APk, HDk, Hk, Hk, 1, APk,
             (long long)Tk * Hk, HDk,
             (long long)Ak * Hk, HDk,
             (long long)NHk * Tk * APk, (long long)Tk * APk,
             Bk * NHk, NHk, 0.0721687836f, 0);
    // 5) masked softmax (in place, zero pad cols) + head-mean -> align matrix
    hipLaunchKernelGGL(softmax_mean_kernel, dim3((unsigned)BT), dim3(256), 0, stream,
                       ws + AT, amask, amat);
    // 6) ctx = attn @ v  (B from v: n-stride 1, k-stride H; K = APk, pads = 0)
    run_gemm(stream, false, ws + AT, ws + VO, nullptr, nullptr, ws + CTX,
             Tk, HDk, APk, APk, 1, Hk, Hk,
             (long long)NHk * Tk * APk, (long long)Tk * APk,
             (long long)Ak * Hk, HDk,
             (long long)Tk * Hk, HDk,
             Bk * NHk, NHk, 1.f, 0);
    // 7) attn_out = ctx @ out_w^T + out_b
    run_gemm(stream, true, ws + CTX, outw, outb, nullptr, ws + AO, (int)BT, Hk, Hk,
             Hk, Hk, 1, Hk, 0,0,0,0,0,0, 1,1, 1.f, 0);
    // 8) ypre = text + attn_out @ Wo^T + bo  (residual fused)
    run_gemm(stream, true, ws + AO, Wo, bo, text, ws + YP, (int)BT, Hk, Hk,
             Hk, Hk, 1, Hk, 0,0,0,0,0,0, 1,1, 1.f, 1);
    // 9) LayerNorm -> y output
    hipLaunchKernelGGL(layernorm_kernel, dim3((unsigned)BT), dim3(256), 0, stream,
                       ws + YP, lng, lnb, yout);
    // 10) h = relu(y @ Wc1^T + bc1)
    run_gemm(stream, true, yout, Wc1, bc1, nullptr, ws + HB, (int)BT, Hk / 2, Hk,
             Hk, Hk, 1, Hk / 2, 0,0,0,0,0,0, 1,1, 1.f, 2);
    // 11) alignment_scores = (h . Wc2 + bc2) * tmask
    hipLaunchKernelGGL(conf_dot_kernel, dim3((unsigned)(BT / 8)), dim3(256), 0, stream,
                       ws + HB, Wc2, bc2, tmask, ascore);
}
